// PatchMixStereo_19997367730718
// MI455X (gfx1250) — matrix operand order repaired and checked by reading, not compiled, not run on hardware
//
#include <hip/hip_runtime.h>
#include <hip/hip_bf16.h>

typedef __attribute__((ext_vector_type(16))) _Float16 v16h;
typedef __attribute__((ext_vector_type(8)))  _Float16 v8h;
typedef __attribute__((ext_vector_type(8)))  float    v8f;
typedef __attribute__((ext_vector_type(4)))  unsigned v4u;
typedef __attribute__((ext_vector_type(8)))  unsigned v8u;

#define BN_    1024
#define C_     32
#define P_     7
#define G_     8
#define DMAX   160
#define W_     167          // P + MAX_DISP
#define TH_    16
#define KTOT   (C_*P_*P_)   // 1568
#define NCHUNK (KTOT/32)    // 49
#define RSTRIP (C_*P_*W_)   // 37408 floats per bn
#define LSIZE  (C_*P_*P_)   // 1568 floats per bn

// CDNA5 async global->LDS copy (ASYNCcnt-tracked, no VGPR round trip).
// LDS dest offset = low 32 bits of the flat shared-aperture address.
__device__ __forceinline__ void async_copy_b128(void* lds_ptr, const void* gptr) {
  unsigned loff = (unsigned)(unsigned long long)lds_ptr;
  asm volatile("global_load_async_to_lds_b128 %0, %1, off"
               :: "v"(loff), "v"((unsigned long long)gptr) : "memory");
}
__device__ __forceinline__ void wait_async0() {
  asm volatile("s_wait_asynccnt 0x0" ::: "memory");
}

// CDNA5 Tensor Data Mover: 1D contiguous f32 tile, global -> LDS.
// D# group0: count=1 | lds_addr | global_addr[56:0] | type=2.
// D# group1: data_size=4B, tensor_dim0 = tile_dim0 = nelem, dims1 = 1.
// One instruction DMAs the whole tile; completion via TENSORcnt.
__device__ __forceinline__ void tdm_load_1d_f32(void* lds_ptr, const void* gptr,
                                                unsigned nelem) {
  unsigned long long va = (unsigned long long)gptr;
  v4u g0;
  g0[0] = 1u;                                            // count=1, user mode
  g0[1] = (unsigned)(unsigned long long)lds_ptr;         // lds_addr (bytes)
  g0[2] = (unsigned)(va & 0xFFFFFFFFull);                // global_addr[31:0]
  g0[3] = (unsigned)((va >> 32) & 0x01FFFFFFull) | 0x80000000u; // [56:32] | type=2
  v8u g1;
  g1[0] = 0x00020000u;                                   // wg_mask=0, data_size=4B
  g1[1] = (nelem & 0xFFFFu) << 16;                       // tensor_dim0[15:0]
  g1[2] = ((nelem >> 16) & 0xFFFFu) | (1u << 16);        // dim0[31:16] | tensor_dim1=1
  g1[3] = (nelem & 0xFFFFu) << 16;                       // tile_dim0 (<=65535)
  g1[4] = 1u;                                            // tile_dim1=1, tile_dim2=0
  g1[5] = nelem;                                         // tensor_dim0_stride[31:0]
  g1[6] = 0u;                                            // stride0 hi | stride1 lo
  g1[7] = 0u;
  asm volatile("tensor_load_to_lds %0, %1" :: "s"(g0), "s"(g1) : "memory");
}

// k = c*49 + y*7 + x  ->  LDS float index c*1169 + y*167 + x = k + 160*(k/7)
__device__ __forceinline__ int strip_addr(int k) { return k + 160 * (k / 7); }

// ---------------------------------------------------------------------------
// Kernel 1: group-wise correlation cost volume via WMMA.
// Per block (one bn): TDM-DMA the full right strip (150 KB) into LDS while the
// waves async-stage the left patch and build packed f16 B fragments, then 10
// waves each accumulate one 16-disparity tile:
//   D[16x16] += A[16x32] * B[32x16] over 49 K-chunks,
//   A[d,k] = right[c,y,sl-d+x] gathered from LDS, B[k,g] = left-block-diag.
// ---------------------------------------------------------------------------
__global__ void __launch_bounds__(320)
cv_wmma_kernel(const float* __restrict__ left, const float* __restrict__ right,
               const int* __restrict__ sl_arr, float* __restrict__ feature)
{
  extern __shared__ char smem[];
  float*    sR = (float*)smem;                 // RSTRIP floats
  _Float16* sB = (_Float16*)(sR + RSTRIP);     // NCHUNK*32*16 packed B frags
  float*    sL = (float*)(sB + NCHUNK*32*16);  // LSIZE floats

  const int bn   = blockIdx.x;
  const int tid  = threadIdx.x;
  const int lane = tid & 31;
  const int wave = tid >> 5;
  const int hi   = (lane >> 4) & 1;            // K-half selector per WMMA layout
  const int lr   = lane & 15;

  // ---- wave 0: single TDM op for the 150 KB right strip ----
  if (wave == 0)
    tdm_load_1d_f32(sR, right + (size_t)bn * RSTRIP, RSTRIP);

  // ---- all waves: async-stage left patch (392 x 16B) ----
  const float* lsrc = left + (size_t)bn * LSIZE;
  for (int i = tid; i < LSIZE / 4; i += 320)
    async_copy_b128(sL + i * 4, lsrc + i * 4);
  wait_async0();
  __syncthreads();                             // sL visible

  // ---- overlap with TDM: packed B frags sB[kc][lane][s], K = 8*hi + (s&7) + 16*(s>>3) ----
  for (int idx = tid; idx < NCHUNK * 32; idx += 320) {
    int kc  = idx >> 5;
    int ln  = idx & 31;
    int bhi = (ln >> 4) & 1;
    int col = ln & 15;                         // B column N = group
    _Float16* dst = sB + idx * 16;
#pragma unroll
    for (int s = 0; s < 16; ++s) {
      int kk  = kc * 32 + 8 * bhi + (s & 7) + ((s >> 3) << 4);
      int grp = kk / 196;                      // channel group of this k
      float v = (grp == col) ? sL[kk] : 0.f;
      dst[s] = (_Float16)v;
    }
  }
  if (wave == 0) __builtin_amdgcn_s_wait_tensorcnt(0);
  __syncthreads();                             // sR visible

  const int sl    = sl_arr[bn];
  const int dbase = wave * 16;
  const int dA    = dbase + lr;                // A-matrix row owned by this lane
  int cb = sl - dA;                            // window start column
  cb = cb < 0 ? 0 : (cb > (W_ - P_) ? (W_ - P_) : cb);  // clamp (invalid d zeroed later)
  const int kbase0 = 8 * hi;                   // lane's K-offset inside each chunk

  v8f acc = {0.f,0.f,0.f,0.f,0.f,0.f,0.f,0.f};

  for (int kc = 0; kc < NCHUNK; ++kc) {
    const int kb = kc * 32 + kbase0;
    v16h a;
    // A layout (16-bit A 16x32): lane<16 holds K{0..7,16..23}, lane>=16 K{8..15,24..31}
#pragma unroll
    for (int h = 0; h < 16; ++h) {
      int k = kb + (h & 7) + ((h >> 3) << 4);  // off(h) in {0..7,16..23}
      a[h] = (_Float16)sR[strip_addr(k) + cb];
    }
    const v8h* bp = (const v8h*)(sB + (kc * 32 + lane) * 16);
    v8h b0 = bp[0], b1 = bp[1];
    v16h b = __builtin_shufflevector(b0, b1, 0,1,2,3,4,5,6,7,8,9,10,11,12,13,14,15);
    acc = __builtin_amdgcn_wmma_f32_16x16x32_f16(false, a, false, b,
                                                 (short)0, acc, false, false);
  }

  // D layout: lane = column N (group), vgpr j -> row M = j + 8*hi (disparity)
  if (lr < G_) {
    float* fout = feature + (size_t)bn * (DMAX * G_);
#pragma unroll
    for (int j = 0; j < 8; ++j) {
      int d = dbase + j + 8 * hi;
      float v = (d <= sl) ? acc[j] * (1.0f / 196.0f) : 0.0f;  // valid mask
      fout[d * G_ + lr] = v;
    }
  }
}

// ---------------------------------------------------------------------------
// Kernel 2: one block per bn. WMMA Gram matrix -> kNN(3) -> hypergraph
// pos/neg mean-agg conv -> conv1d(k=3) -> masked softmax -> disparity.
// ---------------------------------------------------------------------------
__global__ void __launch_bounds__(256)
hgnn_kernel(const float* __restrict__ feature,
            const float* __restrict__ f1w, const float* __restrict__ f1b,
            const float* __restrict__ cw,  const float* __restrict__ cbias,
            float* __restrict__ out)
{
  extern __shared__ char smem2[];
  float* sF    = (float*)smem2;        // [160][8] original features (vol^T)
  float* sH    = sF    + 1280;         // [160][8] leakyrelu(linear)
  float* sE0   = sH    + 1280;         // [160][8] pos-branch edge means
  float* sE1   = sE0   + 1280;
  float* sA0   = sE1   + 1280;         // scatter accumulators
  float* sA1   = sA0   + 1280;
  float* sFeat = sA1   + 1280;         // [160][8] refined features
  float* sGram = sFeat + 1280;         // [160][160]
  float* sNorm = sGram + 25600;        // [160]
  float* sC0   = sNorm + 160;          // scatter counts
  float* sC1   = sC0   + 160;
  float* sAgg  = sC1   + 160;          // conv1d output
  float* sRed  = sAgg  + 160;          // [256] rowsum stash / reductions
  _Float16* sFh = (_Float16*)(sRed + 256);   // [160][8] f16 features
  int* sIdxN   = (int*)(sFh + 1280);   // [160][3] idx | (near<<16)

  const int bn   = blockIdx.x;
  const int tid  = threadIdx.x;
  const int lane = tid & 31;
  const int wave = tid >> 5;
  const int hi   = (lane >> 4) & 1;
  const int lr   = lane & 15;

  if (wave == 0) {
    tdm_load_1d_f32(sF, feature + (size_t)bn * (DMAX * G_), DMAX * G_);
    __builtin_amdgcn_s_wait_tensorcnt(0);
  }
  __syncthreads();

  for (int i = tid; i < DMAX * G_; i += 256) sFh[i] = (_Float16)sF[i];
  if (tid < DMAX) {                    // norms + rowsum (mask) per node
    float n = 0.f, rs = 0.f;
#pragma unroll
    for (int g = 0; g < G_; ++g) { float v = sF[tid*G_+g]; n += v*v; rs += v; }
    sNorm[tid] = n;
    sRed[tid]  = rs;                   // stashed until conv step
  }
  __syncthreads();

  // Gram = F * F^T via WMMA (K=8 zero-padded to 32), 10x10 tiles over 8 waves
  for (int t = wave; t < 100; t += 8) {
    int i0 = (t / 10) * 16, j0 = (t % 10) * 16;
    int irow = i0 + lr, jcol = j0 + lr;
    bool lo = (hi == 0);
    v16h a, b;
#pragma unroll
    for (int h = 0; h < 16; ++h)
      a[h] = (lo && h < 8) ? sFh[irow*8 + h] : (_Float16)0.f;
#pragma unroll
    for (int s = 0; s < 16; ++s)
      b[s] = (lo && s < 8) ? sFh[jcol*8 + s] : (_Float16)0.f;
    v8f c0 = {0.f,0.f,0.f,0.f,0.f,0.f,0.f,0.f};
    c0 = __builtin_amdgcn_wmma_f32_16x16x32_f16(false, a, false, b,
                                                (short)0, c0, false, false);
#pragma unroll
    for (int j = 0; j < 8; ++j)
      sGram[(i0 + j + 8*hi) * DMAX + jcol] = c0[j];
  }
  __syncthreads();

  // stable top-3 smallest distances (ties -> lower index, as lax.top_k)
  if (tid < DMAX) {
    const int i = tid;
    float ni = sNorm[i];
    float b0 = 3.4e38f, b1 = 3.4e38f, b2 = 3.4e38f;
    int i0n = 0, i1n = 0, i2n = 0;
    for (int j = 0; j < DMAX; ++j) {
      float dv = ni + sNorm[j] - 2.f * sGram[i*DMAX + j];
      if      (dv < b0) { b2=b1; i2n=i1n; b1=b0; i1n=i0n; b0=dv; i0n=j; }
      else if (dv < b1) { b2=b1; i2n=i1n; b1=dv; i1n=j; }
      else if (dv < b2) { b2=dv; i2n=j; }
    }
    int ids[3] = {i0n, i1n, i2n};
#pragma unroll
    for (int k = 0; k < 3; ++k) {
      int idn  = ids[k];
      int diff = idn - i; if (diff < 0) diff = -diff;
      sIdxN[i*3 + k] = idn | ((diff < TH_) ? (1 << 16) : 0);
    }
    // h = leaky_relu(x @ W^T + b, 0.01)
    float x[8];
#pragma unroll
    for (int g = 0; g < 8; ++g) x[g] = sF[i*8+g];
#pragma unroll
    for (int go = 0; go < 8; ++go) {
      float acc = f1b[go];
#pragma unroll
      for (int gi = 0; gi < 8; ++gi) acc += x[gi] * f1w[go*8+gi];
      sH[i*8+go] = (acc > 0.f) ? acc : 0.01f * acc;
    }
  }
  for (int i = tid; i < DMAX*G_; i += 256) { sA0[i] = 0.f; sA1[i] = 0.f; }
  if (tid < DMAX) { sC0[tid] = 0.f; sC1[tid] = 0.f; }
  __syncthreads();

  // E = mean of h over (near / far) neighbors of each node
  if (tid < DMAX) {
    const int i = tid;
    float ap[8] = {0,0,0,0,0,0,0,0}, an[8] = {0,0,0,0,0,0,0,0};
    int cp = 0, cn = 0;
#pragma unroll
    for (int k = 0; k < 3; ++k) {
      int e = sIdxN[i*3+k]; int idn = e & 0xFFFF; int nr = e >> 16;
      if (nr) { cp++;
#pragma unroll
        for (int g = 0; g < 8; ++g) ap[g] += sH[idn*8+g];
      } else { cn++;
#pragma unroll
        for (int g = 0; g < 8; ++g) an[g] += sH[idn*8+g];
      }
    }
    float rp = cp ? 1.f/(float)cp : 0.f, rn = cn ? 1.f/(float)cn : 0.f;
#pragma unroll
    for (int g = 0; g < 8; ++g) { sE0[i*8+g] = ap[g]*rp; sE1[i*8+g] = an[g]*rn; }
  }
  __syncthreads();

  // transpose direction: scatter E[c] to rows r = idx[c,k] (ds_add_f32 atomics)
  if (tid < DMAX) {
    const int c = tid;
#pragma unroll
    for (int k = 0; k < 3; ++k) {
      int e = sIdxN[c*3+k]; int idn = e & 0xFFFF; int nr = e >> 16;
      if (nr) {
        atomicAdd(&sC0[idn], 1.f);
#pragma unroll
        for (int g = 0; g < 8; ++g) atomicAdd(&sA0[idn*8+g], sE0[c*8+g]);
      } else {
        atomicAdd(&sC1[idn], 1.f);
#pragma unroll
        for (int g = 0; g < 8; ++g) atomicAdd(&sA1[idn*8+g], sE1[c*8+g]);
      }
    }
  }
  __syncthreads();

  if (tid < DMAX) {                    // feat = feature + 0.1*x1 - 0.1*x2
    const int r = tid;
    float c0n = sC0[r], c1n = sC1[r];
    float r0 = c0n > 0.f ? 1.f/c0n : 0.f;
    float r1 = c1n > 0.f ? 1.f/c1n : 0.f;
#pragma unroll
    for (int g = 0; g < 8; ++g)
      sFeat[r*8+g] = sF[r*8+g] + 0.1f * sA0[r*8+g] * r0 - 0.1f * sA1[r*8+g] * r1;
  }
  __syncthreads();

  if (tid < DMAX) {                    // conv1d k=3 pad=1 over D + mask
    const int d = tid;
    float a = cbias[0];
#pragma unroll
    for (int g = 0; g < 8; ++g) {
      float fm = (d > 0)        ? sFeat[(d-1)*8+g] : 0.f;
      float f0 =                  sFeat[d*8+g];
      float fp = (d < DMAX-1)   ? sFeat[(d+1)*8+g] : 0.f;
      a += cw[g*3+0]*fm + cw[g*3+1]*f0 + cw[g*3+2]*fp;
    }
    if (sRed[d] == 0.f) a = -1e9f;     // sum_g vol[g,d] == 0 -> NEG
    sAgg[d] = a;
  }
  __syncthreads();

  // softmax over D + disparity regression
  float av = (tid < DMAX) ? sAgg[tid] : -3.4e38f;
  sRed[tid] = av;
  __syncthreads();
  for (int s = 128; s > 0; s >>= 1) {
    if (tid < s) sRed[tid] = fmaxf(sRed[tid], sRed[tid+s]);
    __syncthreads();
  }
  float mx = sRed[0];
  __syncthreads();
  float ev = (tid < DMAX) ? __expf(av - mx) : 0.f;
  sRed[tid] = ev;
  __syncthreads();
  for (int s = 128; s > 0; s >>= 1) {
    if (tid < s) sRed[tid] += sRed[tid+s];
    __syncthreads();
  }
  float ssum = sRed[0];
  __syncthreads();
  sRed[tid] = ev * (float)tid;
  __syncthreads();
  for (int s = 128; s > 0; s >>= 1) {
    if (tid < s) sRed[tid] += sRed[tid+s];
    __syncthreads();
  }
  if (tid == 0) out[bn] = sRed[0] / ssum;
}

extern "C" void kernel_launch(void* const* d_in, const int* in_sizes, int n_in,
                              void* d_out, int out_size, void* d_ws, size_t ws_size,
                              hipStream_t stream) {
  (void)in_sizes; (void)n_in; (void)out_size; (void)ws_size;
  const float* left  = (const float*)d_in[0];
  const float* right = (const float*)d_in[1];
  const int*   sl    = (const int*)d_in[2];
  const float* f1w   = (const float*)d_in[3];
  const float* f1b   = (const float*)d_in[4];
  const float* cw    = (const float*)d_in[5];
  const float* cb    = (const float*)d_in[6];
  float* out     = (float*)d_out;
  float* featbuf = (float*)d_ws;   // BN*D*G floats = 5.24 MB scratch

  size_t sh1 = (size_t)RSTRIP * sizeof(float)              // right strip f32
             + (size_t)NCHUNK * 32 * 16 * sizeof(_Float16) // packed B frags
             + (size_t)LSIZE * sizeof(float);              // left patch
  size_t sh2 = (size_t)35456 * sizeof(float) + 1280 * sizeof(_Float16) + 480 * sizeof(int);

  cv_wmma_kernel<<<dim3(BN_), dim3(320), sh1, stream>>>(left, right, sl, featbuf);
  hgnn_kernel  <<<dim3(BN_), dim3(256), sh2, stream>>>(featbuf, f1w, f1b, cw, cb, out);
}
